// GRUCell_53987738911237
// MI455X (gfx1250) — matrix operand order repaired and checked by reading, not compiled, not run on hardware
//
#include <hip/hip_runtime.h>

#define BATCH   1024
#define ISZ     256
#define DDIM    128
#define HSZ     (ISZ * DDIM)          // 32768
#define MT      32                    // batch rows per workgroup
#define UELEM   ((size_t)ISZ * DDIM * DDIM)   // 4,194,304 elems per U
#define WS_NEED (3 * UELEM * 2)       // 25,165,824 bytes of bf16 U

typedef __attribute__((ext_vector_type(16))) __bf16 v16bf;
typedef __attribute__((ext_vector_type(8)))  float  v8f;

union FragAB {
    v16bf          v;
    unsigned short us[16];
    unsigned int   u[8];
    uint4          q[2];
};

__device__ __forceinline__ unsigned short f2bf(float f) {
    unsigned u = __builtin_bit_cast(unsigned, f);
    u += 0x7FFFu + ((u >> 16) & 1u);          // round-to-nearest-even
    return (unsigned short)(u >> 16);
}

__device__ __forceinline__ float sigmoidf(float x) {
    return 1.0f / (1.0f + __expf(-x));
}

// ---------------- U f32 -> bf16 pre-conversion ----------------
__global__ __launch_bounds__(256) void convert_u(
    const float* __restrict__ Ur, const float* __restrict__ Uz,
    const float* __restrict__ Uh, unsigned short* __restrict__ ws)
{
    const float* U = (blockIdx.y == 0) ? Ur : (blockIdx.y == 1) ? Uz : Uh;
    unsigned short* dst = ws + (size_t)blockIdx.y * UELEM;
    const size_t base = ((size_t)blockIdx.x * 256 + threadIdx.x) * 8;
    float4 a = *(const float4*)(U + base);
    float4 b = *(const float4*)(U + base + 4);
    union { unsigned short us[8]; uint4 q; } t;
    t.us[0] = f2bf(a.x); t.us[1] = f2bf(a.y); t.us[2] = f2bf(a.z); t.us[3] = f2bf(a.w);
    t.us[4] = f2bf(b.x); t.us[5] = f2bf(b.y); t.us[6] = f2bf(b.z); t.us[7] = f2bf(b.w);
    *(uint4*)(dst + base) = t.q;
}

// ---------------- B-fragment loaders (32x16 K x N) ----------------
// Lane L holds N column (n0 + L%16); 16 K values in table order (kb+0..7, kb+16..23, kb = kk*32 + (L/16)*8).
__device__ __forceinline__ v16bf loadBfrag(const unsigned short* __restrict__ Ub,
                                           int i, int kk, int n0, int lane) {
    const int n  = n0 + (lane & 15);
    const int kb = kk * 32 + (lane >> 4) * 8;
    FragAB f;
#pragma unroll
    for (int s = 0; s < 16; ++s) {
        const int k = kb + (s < 8 ? s : 8 + s);
        f.us[s] = Ub[((size_t)i * DDIM + k) * DDIM + n];
    }
    return f.v;
}

__device__ __forceinline__ v16bf loadBfrag(const float* __restrict__ U,
                                           int i, int kk, int n0, int lane) {
    const int n  = n0 + (lane & 15);
    const int kb = kk * 32 + (lane >> 4) * 8;
    FragAB f;
#pragma unroll
    for (int s = 0; s < 16; ++s) {
        const int k = kb + (s < 8 ? s : 8 + s);
        f.us[s] = f2bf(U[((size_t)i * DDIM + k) * DDIM + n]);
    }
    return f.v;
}

// A-fragment (16x32 M x K) gathered from a row-major bf16 LDS tile (stride 128).
// Lanes 0-15: rows M=0..15, K pairs {0,1}{2,3}{4,5}{6,7}{16,17}{18,19}{20,21}{22,23};
// lanes 16-31: same rows, K pairs shifted by +8.
__device__ __forceinline__ v16bf loadAfrag(const unsigned short* __restrict__ tile,
                                           int kk, int lane) {
    const unsigned* t32 = (const unsigned*)tile;
    const int m  = lane & 15;
    const int kb = (lane >> 4) * 8;
    FragAB f;
#pragma unroll
    for (int j = 0; j < 4; ++j) {
        int k = kk * 32 + kb + 2 * j;
        f.u[j] = t32[(m * DDIM + k) >> 1];
    }
#pragma unroll
    for (int j = 0; j < 4; ++j) {
        int k = kk * 32 + 16 + kb + 2 * j;
        f.u[4 + j] = t32[(m * DDIM + k) >> 1];
    }
    return f.v;
}

// ---------------- fused grouped-GRU kernel ----------------
template <typename UT>
__global__ __launch_bounds__(256) void gru_group_wmma(
    const float* __restrict__ X,  const float* __restrict__ h,
    const float* __restrict__ Wr, const float* __restrict__ Wz, const float* __restrict__ Wh,
    const UT*    __restrict__ Ur, const UT*    __restrict__ Uz, const UT*    __restrict__ Uh,
    const float* __restrict__ br, const float* __restrict__ bz, const float* __restrict__ bh,
    float* __restrict__ out)
{
    __shared__ float          sHg[MT][DDIM];    // f32 h tile (epilogue)
    __shared__ unsigned short sHgBf[MT][DDIM];  // bf16 h tile (A for GEMMs 1&2)
    __shared__ unsigned short sRh[MT][DDIM];    // bf16 r*h tile (A for GEMM 3)
    __shared__ float          sX[MT];

    const int tid  = threadIdx.x;
    const int wave = tid >> 5;
    const int lane = tid & 31;
    const int m0   = blockIdx.x * MT;   // batch-row tile
    const int i    = blockIdx.y;        // group index
    const int n0   = wave * 16;         // this wave's N-tile (D columns)

    // ---- stage h tile: 256 threads x 16 contiguous floats = 32x128 ----
    {
        const int row = tid >> 3;
        const int c0  = (tid & 7) * 16;
        const float4* p =
            (const float4*)(h + (size_t)(m0 + row) * HSZ + i * DDIM + c0);
#pragma unroll
        for (int q = 0; q < 4; ++q) {
            float4 a = p[q];
            const int c = c0 + q * 4;
            sHg[row][c + 0] = a.x; sHg[row][c + 1] = a.y;
            sHg[row][c + 2] = a.z; sHg[row][c + 3] = a.w;
            sHgBf[row][c + 0] = f2bf(a.x); sHgBf[row][c + 1] = f2bf(a.y);
            sHgBf[row][c + 2] = f2bf(a.z); sHgBf[row][c + 3] = f2bf(a.w);
        }
    }
    if (tid < MT) sX[tid] = X[(size_t)(m0 + tid) * ISZ + i];
    __syncthreads();

    // ---- GEMM 1 (h @ U_r) and GEMM 2 (h @ U_z), 2 M-subtiles each ----
    v8f racc[2] = {{}, {}};
    v8f zacc[2] = {{}, {}};
    {
        v16bf bfr[4];
#pragma unroll
        for (int kk = 0; kk < 4; ++kk) bfr[kk] = loadBfrag(Ur, i, kk, n0, lane);
#pragma unroll
        for (int mt = 0; mt < 2; ++mt)
#pragma unroll
            for (int kk = 0; kk < 4; ++kk)
                racc[mt] = __builtin_amdgcn_wmma_f32_16x16x32_bf16(
                    false, loadAfrag(&sHgBf[mt * 16][0], kk, lane),
                    false, bfr[kk], (short)0, racc[mt], false, false);
#pragma unroll
        for (int kk = 0; kk < 4; ++kk) bfr[kk] = loadBfrag(Uz, i, kk, n0, lane);
#pragma unroll
        for (int mt = 0; mt < 2; ++mt)
#pragma unroll
            for (int kk = 0; kk < 4; ++kk)
                zacc[mt] = __builtin_amdgcn_wmma_f32_16x16x32_bf16(
                    false, loadAfrag(&sHgBf[mt * 16][0], kk, lane),
                    false, bfr[kk], (short)0, zacc[mt], false, false);
    }

    // ---- elementwise r, z for this wave's own 16x16 tiles ----
    // C/D layout: lane&15 = N column; lanes>=16 hold rows M=8..15; vgpr j = row.
    const int nl = lane & 15;
    const int d  = n0 + nl;
    const int mb = (lane >> 4) * 8;
    const float wrv = Wr[(size_t)i * DDIM + d];
    const float wzv = Wz[(size_t)i * DDIM + d];
    const float brv = br[(size_t)i * DDIM + d];
    const float bzv = bz[(size_t)i * DDIM + d];

    float zv[16];
#pragma unroll
    for (int mt = 0; mt < 2; ++mt)
#pragma unroll
        for (int j = 0; j < 8; ++j) {
            const int m    = mt * 16 + mb + j;
            const float xv = sX[m];
            const float r  = sigmoidf(racc[mt][j] + xv * wrv + brv);
            zv[mt * 8 + j] = sigmoidf(zacc[mt][j] + xv * wzv + bzv);
            sRh[m][d] = f2bf(r * sHg[m][d]);
        }
    __syncthreads();

    // ---- GEMM 3: (r*h) @ U_h ----
    v8f hacc[2] = {{}, {}};
    {
        v16bf bfr[4];
#pragma unroll
        for (int kk = 0; kk < 4; ++kk) bfr[kk] = loadBfrag(Uh, i, kk, n0, lane);
#pragma unroll
        for (int mt = 0; mt < 2; ++mt)
#pragma unroll
            for (int kk = 0; kk < 4; ++kk)
                hacc[mt] = __builtin_amdgcn_wmma_f32_16x16x32_bf16(
                    false, loadAfrag(&sRh[mt * 16][0], kk, lane),
                    false, bfr[kk], (short)0, hacc[mt], false, false);
    }

    // ---- epilogue: h_tilde, h_new ----
    const float whv = Wh[(size_t)i * DDIM + d];
    const float bhv = bh[(size_t)i * DDIM + d];
#pragma unroll
    for (int mt = 0; mt < 2; ++mt)
#pragma unroll
        for (int j = 0; j < 8; ++j) {
            const int m     = mt * 16 + mb + j;
            const float ht  = tanhf(hacc[mt][j] + sX[m] * whv + bhv);
            const float hgv = sHg[m][d];
            const float z   = zv[mt * 8 + j];
            const float hn  = z * hgv + (1.0f - z) * ht;
            const size_t off = (size_t)(m0 + m) * HSZ + (size_t)i * DDIM + d;
            out[off]                       = hn;  // h_new
            out[(size_t)BATCH * HSZ + off] = ht;  // h_tilde
        }
}

extern "C" void kernel_launch(void* const* d_in, const int* in_sizes, int n_in,
                              void* d_out, int out_size, void* d_ws, size_t ws_size,
                              hipStream_t stream) {
    (void)in_sizes; (void)n_in; (void)out_size;
    const float* X  = (const float*)d_in[0];
    const float* h  = (const float*)d_in[1];
    const float* Wr = (const float*)d_in[2];
    const float* Wz = (const float*)d_in[3];
    const float* Wh = (const float*)d_in[4];
    const float* Ur = (const float*)d_in[5];
    const float* Uz = (const float*)d_in[6];
    const float* Uh = (const float*)d_in[7];
    const float* br = (const float*)d_in[8];
    const float* bz = (const float*)d_in[9];
    const float* bh = (const float*)d_in[10];
    float* out = (float*)d_out;

    dim3 grid(BATCH / MT, ISZ);   // 32 M-tiles x 256 groups
    dim3 block(256);              // 8 waves; wave w owns N-tile w

    if (ws_size >= WS_NEED) {
        // Pre-convert U to bf16 once per launch; main kernel loads packed bf16.
        unsigned short* Ub = (unsigned short*)d_ws;
        dim3 cgrid((unsigned)(UELEM / (256 * 8)), 3);
        convert_u<<<cgrid, 256, 0, stream>>>(Ur, Uz, Uh, Ub);
        gru_group_wmma<unsigned short><<<grid, block, 0, stream>>>(
            X, h, Wr, Wz, Wh,
            Ub, Ub + UELEM, Ub + 2 * UELEM,
            br, bz, bh, out);
    } else {
        // Fallback: inline f32->bf16 conversion of U fragments.
        gru_group_wmma<float><<<grid, block, 0, stream>>>(
            X, h, Wr, Wz, Wh, Ur, Uz, Uh, br, bz, bh, out);
    }
}
